// LSTM_64003602645117
// MI455X (gfx1250) — matrix operand order repaired and checked by reading, not compiled, not run on hardware
//
#include <hip/hip_runtime.h>
#include <hip/hip_bf16.h>

// ---------------------------------------------------------------------------
// LSTM forward for MI455X (gfx1250), wave32, WMMA 16x16x32 bf16 (fp32 accum).
//
// Phase 1: gx = x @ W   (full-chip WMMA GEMM, 131072 x 1024 x 256), results
//          stored in C-tile layout (per-lane v8f) so the scan re-reads them
//          with zero shuffling, fully coalesced.
// Phase 2: sequential scan, 4 persistent workgroups (one per 16 batch rows),
//          all sync workgroup-local. U is loop-invariant: kt 0..3 in LDS
//          (256KB, loaded once via global_load_async_to_lds_b128), kt 4..7 in
//          per-wave VGPRs -> zero per-step global traffic for U. Gates cross
//          LDS bank-swizzled so the elementwise phase does b128 ds loads.
// ---------------------------------------------------------------------------

typedef __attribute__((ext_vector_type(16))) __bf16          v16bf;
typedef __attribute__((ext_vector_type(8)))  float           v8f;
typedef __attribute__((ext_vector_type(8)))  unsigned short  v8u16;

#define T_STEPS 2048
#define BATCH   64
#define DIM     256   // D == H
#define G4      1024  // 4*H

// LDS partition (ushorts): U half (kt 0..3) | gates bf16 | h A-tiles
#define ULDS_USHORTS   (4 * 64 * 32 * 16)   // 131072 ushorts = 256KB
#define GATES_USHORTS  (16 * G4)            // 16384 ushorts  = 32KB
#define HA_USHORTS     (8 * 32 * 16)        // 4096 ushorts   = 8KB
#define SMEM_BYTES     ((ULDS_USHORTS + GATES_USHORTS + HA_USHORTS) * 2)

__device__ __forceinline__ unsigned short f2bf(float f) {
    return __builtin_bit_cast(unsigned short, (__bf16)f);   // RNE
}
__device__ __forceinline__ float bf2f(unsigned short u) {
    return __builtin_bit_cast(float, (unsigned int)u << 16); // exact
}
__device__ __forceinline__ float sigmoidf_(float x) {
    return 1.0f / (1.0f + __expf(-x));
}
// Bank-swizzled gates index: row m, logical column n (0..1023).
// XOR within the 256-wide gate block by m*8 -> the elementwise b128 reads
// (16 rows, same 8-col group) land on 16 distinct 16B offsets mod 256B.
__device__ __forceinline__ int gphys(int m, int n) {
    return m * G4 + (n & 768) + ((n & 255) ^ ((m & 15) << 3));
}

// ---------------------------------------------------------------------------
// Pack one [256, 1024] fp32 weight (4 gate matrices [256,256], gate-major in
// N) into bf16 B-operand tiles. Tile (kt, nt) covers K = kt*32..+31,
// N = nt*16..+15. Per-lane layout (ISA 7.12.2, 16-bit B 32x16):
//   lane L: N = nt*16 + (L&15);  K = kt*32 + 8*(L>>4) + (e&7) + 16*(e>>3),  e = 0..15
// ---------------------------------------------------------------------------
__global__ __launch_bounds__(256) void pack_weights(
    const float* __restrict__ W0, const float* __restrict__ W1,
    const float* __restrict__ W2, const float* __restrict__ W3,
    unsigned short* __restrict__ dst)
{
    const int idx = blockIdx.x * 256 + threadIdx.x;   // 0 .. 262143
    const int e    = idx & 15;
    const int L    = (idx >> 4) & 31;
    const int tile = idx >> 9;          // 0..511
    const int kt   = tile >> 6;         // 0..7
    const int nt   = tile & 63;         // 0..63
    const int n    = nt * 16 + (L & 15);
    const int k    = kt * 32 + 8 * (L >> 4) + (e & 7) + 16 * (e >> 3);
    const float* Ws[4] = {W0, W1, W2, W3};
    const float v = Ws[n >> 8][k * 256 + (n & 255)];
    dst[idx] = f2bf(v);
}

// ---------------------------------------------------------------------------
// gx = x @ W, stored in C-tile layout:
//   gx[(((mtile*16 + nslice)*4 + nt) * 256) + lane*8 + r]  (fp32)
// One wave computes a [16 M x 64 N] block: 4 accumulators, 8 k-steps.
// ---------------------------------------------------------------------------
__global__ __launch_bounds__(256) void gx_gemm(
    const float* __restrict__ x, const unsigned short* __restrict__ Wc,
    float* __restrict__ gx)
{
    const int lane   = threadIdx.x & 31;
    const int wid    = blockIdx.x * 8 + (threadIdx.x >> 5);
    const int mtile  = wid >> 4;     // 0..8191
    const int nslice = wid & 15;     // 0..15 (64 N-cols each)
    const int lhi    = lane >> 4;
    const int llo    = lane & 15;
    const size_t row = (size_t)mtile * 16 + llo;

    v8f acc[4] = {};

#pragma unroll
    for (int kt = 0; kt < 8; ++kt) {
        const int kbase = kt * 32 + 8 * lhi;
        const float4 fa = *(const float4*)(x + row * DIM + kbase);
        const float4 fb = *(const float4*)(x + row * DIM + kbase + 4);
        const float4 fc = *(const float4*)(x + row * DIM + kbase + 16);
        const float4 fd = *(const float4*)(x + row * DIM + kbase + 20);
        const float f[16] = {fa.x, fa.y, fa.z, fa.w, fb.x, fb.y, fb.z, fb.w,
                             fc.x, fc.y, fc.z, fc.w, fd.x, fd.y, fd.z, fd.w};
        v16bf a;
#pragma unroll
        for (int e = 0; e < 16; ++e) a[e] = (__bf16)f[e];

#pragma unroll
        for (int nt = 0; nt < 4; ++nt) {
            const int ntg = nslice * 4 + nt;
            const v16bf b = *(const v16bf*)(Wc + ((kt * 64 + ntg) * 32 + lane) * 16);
            acc[nt] = __builtin_amdgcn_wmma_f32_16x16x32_bf16(
                false, a, false, b, (short)0, acc[nt], false, false);
        }
    }

#pragma unroll
    for (int nt = 0; nt < 4; ++nt) {
        float* gp = gx + (((size_t)mtile * 16 + nslice) * 4 + nt) * 256 + lane * 8;
        *(v8f*)gp = acc[nt];
    }
}

// ---------------------------------------------------------------------------
// Persistent scan. blockIdx.x = batch group (16 rows). 16 waves x 32 lanes.
// Wave w owns gate columns [64w, 64w+64). Elementwise: lane (w, L) owns row
// m = L&15, columns jbase..jbase+7 with jbase = 16w + 8*(L>>4).
// ---------------------------------------------------------------------------
__global__ __launch_bounds__(512) void lstm_scan(
    const float* __restrict__ gx, const unsigned short* __restrict__ Uc,
    const float* __restrict__ h0, const float* __restrict__ c0,
    const float* __restrict__ b_i, const float* __restrict__ b_f,
    const float* __restrict__ b_c, const float* __restrict__ b_o,
    float* __restrict__ out)
{
    extern __shared__ __align__(32) unsigned char smem_raw[];
    unsigned short* Ulds     = (unsigned short*)smem_raw;       // kt 0..3 tiles
    unsigned short* gatesLDS = Ulds + ULDS_USHORTS;             // swizzled
    unsigned short* hA       = gatesLDS + GATES_USHORTS;        // h A-tiles

    const int bg   = blockIdx.x;        // 0..3
    const int tid  = threadIdx.x;
    const int w    = tid >> 5;          // wave 0..15
    const int lane = tid & 31;
    const int lhi  = lane >> 4;
    const int llo  = lane & 15;

    // --- one-time: U kt=0..3 (256KB) -> LDS via async DMA loads -----------
    {
        const unsigned lds_base = (unsigned)(uintptr_t)(void*)Ulds;
        for (int i = tid; i < ULDS_USHORTS / 8; i += 512) {
            const unsigned lds_off = lds_base + (unsigned)i * 16u;
            const void* gp = (const void*)(Uc + (size_t)i * 8);
            asm volatile("global_load_async_to_lds_b128 %0, %1, off"
                         :: "v"(lds_off), "v"(gp) : "memory");
        }
        asm volatile("s_wait_asynccnt 0" ::: "memory");
    }

    // --- one-time: preload U kt=4..7 tiles for this wave into registers ---
    v16bf Breg[16];
#pragma unroll
    for (int kt2 = 0; kt2 < 4; ++kt2) {
#pragma unroll
        for (int nt = 0; nt < 4; ++nt) {
            const int ntg = w * 4 + nt;
            Breg[kt2 * 4 + nt] =
                *(const v16bf*)(Uc + (((kt2 + 4) * 64 + ntg) * 32 + lane) * 16);
        }
    }

    // --- init h A-tiles from h0 (bf16, native A layout) ---
    if (tid < 256) {
        const int kt = tid >> 5, L = tid & 31;
        const int m = L & 15, Lg = L >> 4;
        unsigned short* dst = &hA[(kt * 32 + L) * 16];
#pragma unroll
        for (int e = 0; e < 16; ++e) {
            const int K = kt * 32 + (e & 7) + 8 * Lg + 16 * (e >> 3);
            dst[e] = f2bf(h0[(bg * 16 + m) * DIM + K]);
        }
    }

    // --- persistent per-lane cell state (elementwise lane mapping) ---
    const int m_el   = llo;                 // owned row
    const int jbase  = w * 16 + 8 * lhi;    // owned 8-column group
    float c_reg[8];
#pragma unroll
    for (int v = 0; v < 8; ++v)
        c_reg[v] = c0[(bg * 16 + m_el) * DIM + jbase + v];

    // --- hoisted bias (constant over t) for this wave's gate slice ---
    const float* bptr[4] = {b_i, b_f, b_c, b_o};
    float biasv[4];
#pragma unroll
    for (int nt = 0; nt < 4; ++nt) {
        const int n = w * 64 + nt * 16 + llo;
        biasv[nt] = bptr[n >> 8][n & 255];
    }

    __syncthreads();

    for (int t = 0; t < T_STEPS; ++t) {
        // prefetch next step's gx block (64KB / WG-step, 128B per thread)
        if (t + 1 < T_STEPS) {
            const float* gxn = gx + ((size_t)(t + 1) * 4 + bg) * 16384;
            __builtin_prefetch(gxn + (size_t)tid * 32, 0, 1);
        }

        // gates = gx_t + bias  (gx already in C-tile layout: v8f per lane)
        v8f acc[4];
        const size_t tb = (((size_t)t * 4 + bg) * 16 + w) * 4;
#pragma unroll
        for (int nt = 0; nt < 4; ++nt) {
            const v8f g = *(const v8f*)(gx + (tb + nt) * 256 + lane * 8);
            acc[nt] = g + biasv[nt];
        }

        // gates += h @ U : kt 0..3 B from LDS, kt 4..7 B from registers
#pragma unroll
        for (int kt = 0; kt < 4; ++kt) {
            const v16bf a = *(const v16bf*)&hA[(kt * 32 + lane) * 16];
#pragma unroll
            for (int nt = 0; nt < 4; ++nt) {
                const int ntg = w * 4 + nt;
                const v16bf b = *(const v16bf*)&Ulds[((kt * 64 + ntg) * 32 + lane) * 16];
                acc[nt] = __builtin_amdgcn_wmma_f32_16x16x32_bf16(
                    false, a, false, b, (short)0, acc[nt], false, false);
            }
        }
#pragma unroll
        for (int kt2 = 0; kt2 < 4; ++kt2) {
            const v16bf a = *(const v16bf*)&hA[((kt2 + 4) * 32 + lane) * 16];
#pragma unroll
            for (int nt = 0; nt < 4; ++nt) {
                acc[nt] = __builtin_amdgcn_wmma_f32_16x16x32_bf16(
                    false, a, false, Breg[kt2 * 4 + nt], (short)0, acc[nt],
                    false, false);
            }
        }

        // exchange gates through LDS (bf16, bank-swizzled)
#pragma unroll
        for (int nt = 0; nt < 4; ++nt) {
            const int n = w * 64 + nt * 16 + llo;
#pragma unroll
            for (int r = 0; r < 8; ++r) {
                const int m = r + 8 * lhi;
                gatesLDS[gphys(m, n)] = f2bf(acc[nt][r]);
            }
        }
        __syncthreads();   // gates visible; all hA reads of this step done

        // elementwise: lane owns (row m_el, cols jbase..jbase+7)
        {
            const int swz = jbase ^ (m_el << 3);
            const v8u16 gi = *(const v8u16*)&gatesLDS[m_el * G4 +   0 + swz];
            const v8u16 gf = *(const v8u16*)&gatesLDS[m_el * G4 + 256 + swz];
            const v8u16 gc = *(const v8u16*)&gatesLDS[m_el * G4 + 512 + swz];
            const v8u16 go = *(const v8u16*)&gatesLDS[m_el * G4 + 768 + swz];
            v8f   hv, cv;
            v8u16 hp;
#pragma unroll
            for (int v = 0; v < 8; ++v) {
                const float i_t = sigmoidf_(bf2f(gi[v]));
                const float f_t = sigmoidf_(bf2f(gf[v]));
                const float ch  = tanhf(bf2f(gc[v]));
                const float o_t = sigmoidf_(bf2f(go[v]));
                const float cn  = f_t * c_reg[v] + i_t * ch;
                const float hn  = o_t * tanhf(cn);
                c_reg[v] = cn;
                hv[v] = hn;
                cv[v] = cn;
                hp[v] = f2bf(hn);
            }
            // outputs[t] : one 32B store per lane
            *(v8f*)&out[((size_t)t * BATCH + bg * 16 + m_el) * DIM + jbase] = hv;

            // h -> A-tile layout for next step's WMMA: one b128 ds store
            const int kt  = jbase >> 5;
            const int Lg  = (jbase >> 3) & 1;
            const int ehi = (jbase >> 4) & 1;
            *(v8u16*)&hA[(kt * 32 + m_el + 16 * Lg) * 16 + 8 * ehi] = hp;

            if (t == T_STEPS - 1) {
                float* hT = out + (size_t)T_STEPS * BATCH * DIM;
                *(v8f*)&hT[(bg * 16 + m_el) * DIM + jbase] = hv;               // h_T
                *(v8f*)&hT[BATCH * DIM + (bg * 16 + m_el) * DIM + jbase] = cv; // c_T
            }
        }
        __syncthreads();   // hA updated for next iteration
    }
}

// ---------------------------------------------------------------------------
// d_in: 0:x 1:h0 2:c0 3..6:W_i,W_f,W_c,W_o 7..10:U_i..U_o 11..14:b_i..b_o
// d_out: outputs[T,B,H] ++ h_T[B,H] ++ c_T[B,H]
// ws: gx fp32 (512MB) | Wc bf16 (512KB) | Uc bf16 (512KB)
// ---------------------------------------------------------------------------
extern "C" void kernel_launch(void* const* d_in, const int* in_sizes, int n_in,
                              void* d_out, int out_size, void* d_ws, size_t ws_size,
                              hipStream_t stream) {
    (void)in_sizes; (void)n_in; (void)out_size; (void)ws_size;
    const float* x  = (const float*)d_in[0];
    const float* h0 = (const float*)d_in[1];
    const float* c0 = (const float*)d_in[2];

    char* ws = (char*)d_ws;
    float* gx = (float*)ws;                                   // 131072*1024*4
    unsigned short* Wc = (unsigned short*)(ws + (size_t)T_STEPS * BATCH * G4 * 4);
    unsigned short* Uc = Wc + 256 * 1024;                     // +512KB

    pack_weights<<<1024, 256, 0, stream>>>(
        (const float*)d_in[3], (const float*)d_in[4],
        (const float*)d_in[5], (const float*)d_in[6], Wc);
    pack_weights<<<1024, 256, 0, stream>>>(
        (const float*)d_in[7], (const float*)d_in[8],
        (const float*)d_in[9], (const float*)d_in[10], Uc);

    gx_gemm<<<16384, 256, 0, stream>>>(x, Wc, gx);

    lstm_scan<<<4, 512, SMEM_BYTES, stream>>>(
        gx, Uc, h0, c0,
        (const float*)d_in[11], (const float*)d_in[12],
        (const float*)d_in[13], (const float*)d_in[14],
        (float*)d_out);
}
